// BiMamba_58085137711712
// MI455X (gfx1250) — matrix operand order repaired and checked by reading, not compiled, not run on hardware
//
#include <hip/hip_runtime.h>
#include <cstdint>

// ---------------------------------------------------------------------------
// BiMamba block for MI455X (gfx1250, wave32, WMMA).
// GEMMs: v_wmma_f32_16x16x32_bf16, f32 accumulate, branch-free K loop
// (all tiles full: N padded to multiples of 128, M multiples of 64).
// Scan: state dim across lanes (16 lanes/row), shfl-xor reduction,
// 131072 threads for the latency-critical sequential stage.
// ---------------------------------------------------------------------------

typedef __bf16 bf16;
typedef __attribute__((ext_vector_type(16))) __bf16 bf16x16;
typedef __attribute__((ext_vector_type(8)))  float  f32x8;

#define D_MODEL 1024
#define D_STATE 16
#define D_CONV  4
#define D_INNER 2048
#define DT_RANK 64
#define BATCH   2
#define SEQ     1024
#define NB      4                 // batch * 2 directions
#define TOK     (NB * SEQ)        // 4096 mamba tokens
#define FTOK    (BATCH * SEQ)     // 2048 ffn tokens
#define D_FF    4096
#define DBL_N   96                // dt(64) + B(16) + C(16)
#define DBL_P   128               // padded stride for dbl / x_proj weight

// ----------------------------- device helpers ------------------------------

__device__ __forceinline__ float act_softplus(float x) {
    return (x > 20.f) ? x : log1pf(expf(x));
}
__device__ __forceinline__ float act_gelu_exact(float x) {
    return 0.5f * x * (1.f + erff(x * 0.70710678118654752f));
}
__device__ __forceinline__ float silu(float x) {
    return x / (1.f + expf(-x));
}

// ----------------------------- tiny converters -----------------------------

__global__ __launch_bounds__(256)
void cvt_f32_bf16(const float* __restrict__ src, bf16* __restrict__ dst, int n) {
    int i = blockIdx.x * 256 + threadIdx.x;
    if (i < n) dst[i] = (bf16)src[i];
}

// x_proj_w [2048,96] f32 -> [2048,128] bf16, zero-padded columns 96..127
__global__ __launch_bounds__(256)
void cvt_pad_xp(const float* __restrict__ src, bf16* __restrict__ dst) {
    int gid = blockIdx.x * 256 + threadIdx.x;          // over D_INNER*DBL_P
    if (gid >= D_INNER * DBL_P) return;
    int r = gid / DBL_P, c = gid % DBL_P;
    dst[gid] = (c < DBL_N) ? (bf16)src[r * DBL_N + c] : (bf16)0.f;
}

// xx = concat([x, flip(x, axis=1)], axis=0)  ->  bf16 [TOK, D_MODEL]
__global__ __launch_bounds__(256)
void build_xx(const float* __restrict__ x, bf16* __restrict__ xx) {
    int gid = blockIdx.x * 256 + threadIdx.x;
    if (gid >= TOK * D_MODEL) return;
    int tok = gid / D_MODEL, d = gid % D_MODEL;
    int nb = tok / SEQ, t = tok % SEQ;
    int src_tok = (nb < BATCH) ? (nb * SEQ + t)
                               : ((nb - BATCH) * SEQ + (SEQ - 1 - t));
    xx[gid] = (bf16)x[(size_t)src_tok * D_MODEL + d];
}

// dt slice of dbl (stride DBL_P) -> bf16 [TOK, DT_RANK]
__global__ __launch_bounds__(256)
void extract_dt(const float* __restrict__ dbl, bf16* __restrict__ dtbf) {
    int gid = blockIdx.x * 256 + threadIdx.x;
    if (gid >= TOK * DT_RANK) return;
    int tok = gid / DT_RANK, j = gid % DT_RANK;
    dtbf[gid] = (bf16)dbl[(size_t)tok * DBL_P + j];
}

// ----------------------------- WMMA GEMM -----------------------------------
// C[M,N] f32 (+ optional bf16 copy) = act(A[M,K]bf16 @ B[K,N]bf16 + bias)
// 256 threads = 8 waves arranged 4(M) x 2(N): block tile 64(M) x 128(N).
// Per wave: 16(M) x 64(N) = 4 WMMA accumulators; branch-free K loop.
// Requires M%64==0, N%128==0, K%32==0 (guaranteed by caller / padding).
//
// Fragment layouts per CDNA5 ISA 7.12.2 (wave32):
//   A 16x32 bf16 : m=lane&15, half=lane>>4; elems 0..7 -> K=half*8+j,
//                  elems 8..15 -> K=16+half*8+(j-8)   (2x 16B loads)
//   B 32x16 bf16 : n=lane&15, elems use A's K order (16 strided 2-byte loads)
//   C/D 16x16 f32: col=lane&15, row=r+8*(lane>>4) for accumulator VGPR r

__global__ __launch_bounds__(256)
void gemm_bf16_wmma(const bf16* __restrict__ A, const bf16* __restrict__ B,
                    float* __restrict__ C, bf16* __restrict__ Cbf,
                    const float* __restrict__ bias,
                    int M, int N, int K, int act)
{
    const int lane = threadIdx.x & 31;
    const int wave = threadIdx.x >> 5;
    const int m0 = blockIdx.y * 64 + (wave >> 1) * 16;
    const int n0 = blockIdx.x * 128 + (wave & 1) * 64;

    const int mrow = lane & 15;
    const int half = lane >> 4;

    f32x8 acc[4] = {};

    const bf16* arow  = A + (size_t)(m0 + mrow) * K + half * 8;
    const bf16* bbase = B + (size_t)(half * 8) * N + n0 + mrow;

    union Frag { bf16x16 v; uint4 q[2]; };

    for (int k = 0; k < K; k += 32) {
        // Unconditional speculative prefetch of next K tile (ISA 10.5:
        // translation failures are silently dropped). Locality 3 -> near.
        __builtin_prefetch((const void*)(arow + k + 32), 0, 3);
        __builtin_prefetch((const void*)(bbase + (size_t)(k + 32) * N), 0, 3);

        Frag af;
        af.q[0] = *(const uint4*)(arow + k);        // K = k + half*8 + 0..7
        af.q[1] = *(const uint4*)(arow + k + 16);   // K = k+16 + half*8 + 0..7

        const bf16* brow = bbase + (size_t)k * N;   // rows K = k + half*8 + {0..7, 16..23}, col n0+j*16+mrow
#pragma unroll
        for (int j = 0; j < 4; ++j) {
            Frag bf;
            for (int s = 0; s < 16; ++s)
                bf.v[s] = brow[(size_t)(s < 8 ? s : s + 8) * N + j * 16];
            acc[j] = __builtin_amdgcn_wmma_f32_16x16x32_bf16(
                false, af.v, false, bf.v, (short)0, acc[j], false, false);
        }
    }

    // Epilogue: bias + activation + f32 (+ optional bf16) store
#pragma unroll
    for (int j = 0; j < 4; ++j) {
        int ncol = n0 + j * 16 + mrow;
        float bv = bias ? bias[ncol] : 0.f;
#pragma unroll
        for (int r = 0; r < 8; ++r) {
            int row = m0 + r + 8 * half;
            float v = acc[j][r] + bv;
            if (act == 1)      v = act_softplus(v);
            else if (act == 2) v = act_gelu_exact(v);
            size_t idx = (size_t)row * N + ncol;
            C[idx] = v;
            if (Cbf) Cbf[idx] = (bf16)v;
        }
    }
}

// --------------------- depthwise conv(4) + SiLU ----------------------------

__global__ __launch_bounds__(256)
void conv_silu(const float* __restrict__ xz, const float* __restrict__ cw,
               const float* __restrict__ cb, float* __restrict__ xc,
               bf16* __restrict__ xcbf)
{
    int gid = blockIdx.x * 256 + threadIdx.x;
    if (gid >= TOK * D_INNER) return;
    int tok = gid / D_INNER, c = gid % D_INNER;
    int nb = tok / SEQ, t = tok % SEQ;
    float s = cb[c];
#pragma unroll
    for (int i = 0; i < D_CONV; ++i) {
        int tt = t - (D_CONV - 1) + i;
        if (tt >= 0)
            s += cw[c * D_CONV + i] *
                 xz[(size_t)(nb * SEQ + tt) * (2 * D_INNER) + c];
    }
    float v = silu(s);
    xc[gid]   = v;
    xcbf[gid] = (bf16)v;
}

// --------------------------- selective scan --------------------------------
// State dimension spread across lanes: each 16-lane group owns one
// (batch-dir, channel) row; lane = state index n. Per timestep: one exp +
// one FMA per lane, then shfl-xor butterfly sum of h*C over the 16 state
// lanes. 131072 threads total (vs 8192 for state-in-registers), and the
// per-step dependency chain shrinks from 16 serial exp+fma to 1 exp + 4
// cross-lane adds.

__global__ __launch_bounds__(256)
void scan_kernel(const float* __restrict__ xz, const float* __restrict__ xc,
                 const float* __restrict__ delta, const float* __restrict__ dbl,
                 const float* __restrict__ A_log, const float* __restrict__ Dp,
                 bf16* __restrict__ ybf)
{
    int gid = blockIdx.x * 256 + threadIdx.x;      // over NB*D_INNER*D_STATE
    int rowid = gid >> 4;                          // (nb, c)
    int n = gid & 15;                              // state index
    if (rowid >= NB * D_INNER) return;
    int nb = rowid / D_INNER, c = rowid % D_INNER;

    float An = -expf(A_log[c * D_STATE + n]);      // A = -exp(A_log)
    float Dc = Dp[c];
    float h = 0.f;

    for (int t = 0; t < SEQ; ++t) {
        int tok = nb * SEQ + t;
        float dlt = delta[(size_t)tok * D_INNER + c];   // broadcast in group
        float u   = xc[(size_t)tok * D_INNER + c];      // broadcast in group
        const float* bc = dbl + (size_t)tok * DBL_P;
        float Bn = bc[DT_RANK + n];
        float Cn = bc[DT_RANK + D_STATE + n];
        h = h * expf(dlt * An) + (dlt * u) * Bn;
        float p = h * Cn;
        p += __shfl_xor(p, 1, 16);
        p += __shfl_xor(p, 2, 16);
        p += __shfl_xor(p, 4, 16);
        p += __shfl_xor(p, 8, 16);
        if (n == 0) {
            float z = xz[(size_t)tok * (2 * D_INNER) + D_INNER + c];
            ybf[(size_t)tok * D_INNER + c] = (bf16)((p + u * Dc) * silu(z));
        }
    }
}

// -------------------- combine directions + LayerNorm -----------------------

__global__ __launch_bounds__(256)
void combine_ln(const float* __restrict__ m, const float* __restrict__ g,
                const float* __restrict__ beta, bf16* __restrict__ hbf)
{
    __shared__ float vals[D_MODEL];
    __shared__ float red[256];
    int row = blockIdx.x;                           // 0 .. FTOK-1
    int b = row / SEQ, t = row % SEQ;
    const float* m1 = m + (size_t)(b * SEQ + t) * D_MODEL;
    const float* m2 = m + (size_t)((BATCH + b) * SEQ + (SEQ - 1 - t)) * D_MODEL;

    float ps = 0.f, ps2 = 0.f;
    for (int i = threadIdx.x; i < D_MODEL; i += 256) {
        float v = m1[i] + m2[i];
        vals[i] = v;
        ps += v; ps2 += v * v;
    }
    red[threadIdx.x] = ps;
    __syncthreads();
    for (int s = 128; s > 0; s >>= 1) {
        if (threadIdx.x < s) red[threadIdx.x] += red[threadIdx.x + s];
        __syncthreads();
    }
    float mu = red[0] * (1.f / D_MODEL);
    __syncthreads();
    red[threadIdx.x] = ps2;
    __syncthreads();
    for (int s = 128; s > 0; s >>= 1) {
        if (threadIdx.x < s) red[threadIdx.x] += red[threadIdx.x + s];
        __syncthreads();
    }
    float var = red[0] * (1.f / D_MODEL) - mu * mu;
    float rstd = rsqrtf(var + 1e-5f);
    for (int i = threadIdx.x; i < D_MODEL; i += 256) {
        float hn = (vals[i] - mu) * rstd * g[i] + beta[i];
        hbf[(size_t)row * D_MODEL + i] = (bf16)hn;
    }
}

// ------------------------------- host side ---------------------------------

extern "C" void kernel_launch(void* const* d_in, const int* in_sizes, int n_in,
                              void* d_out, int out_size, void* d_ws, size_t ws_size,
                              hipStream_t stream)
{
    const float* x      = (const float*)d_in[0];
    const float* w_in   = (const float*)d_in[1];
    const float* conv_w = (const float*)d_in[2];
    const float* conv_b = (const float*)d_in[3];
    const float* w_xp   = (const float*)d_in[4];
    const float* w_dt   = (const float*)d_in[5];
    const float* b_dt   = (const float*)d_in[6];
    const float* A_log  = (const float*)d_in[7];
    const float* Dp     = (const float*)d_in[8];
    const float* w_out  = (const float*)d_in[9];
    const float* ln_g   = (const float*)d_in[10];
    const float* ln_b   = (const float*)d_in[11];
    const float* ff_w1  = (const float*)d_in[12];
    const float* ff_b1  = (const float*)d_in[13];
    const float* ff_w2  = (const float*)d_in[14];
    const float* ff_b2  = (const float*)d_in[15];
    float* out = (float*)d_out;

    char* ws = (char*)d_ws;
    const size_t MB = 1024ull * 1024ull;

    // bf16 weight pool (element offsets)
    bf16* wb = (bf16*)(ws + 0);
    bf16* wb_in  = wb;                                      // 1024*4096
    bf16* wb_xp  = wb_in  + (size_t)D_MODEL * (2 * D_INNER);// 2048*128 (padded)
    bf16* wb_dt  = wb_xp  + (size_t)D_INNER * DBL_P;        // 64*2048
    bf16* wb_out = wb_dt  + (size_t)DT_RANK * D_INNER;      // 2048*1024
    bf16* wb_f1  = wb_out + (size_t)D_INNER * D_MODEL;      // 1024*4096
    bf16* wb_f2  = wb_f1  + (size_t)D_MODEL * D_FF;         // 4096*1024 (< 32 MB)

    bf16*  xxbf  = (bf16*) (ws + 32 * MB);                 //  8 MB [TOK, D_MODEL]
    float* xz    = (float*)(ws + 40 * MB);                 // 64 MB [TOK, 2*D_INNER]
    float* xc    = (float*)(ws + 104 * MB);                // 32 MB [TOK, D_INNER]
    bf16*  xcbf  = (bf16*) (ws + 136 * MB);                // 16 MB
    float* dbl   = (float*)(ws + 152 * MB);                //  2 MB [TOK, DBL_P]
    bf16*  dtbf  = (bf16*) (ws + 154 * MB);                // 0.5 MB [TOK, 64]
    float* delta = (float*)(ws + 155 * MB);                // 32 MB [TOK, D_INNER]
    bf16*  ybf   = (bf16*) (ws + 187 * MB);                // 16 MB [TOK, D_INNER]
    // overlays (producer of previous buffer is dead before reuse):
    float* mbuf  = (float*)(ws + 155 * MB);                // over delta: [TOK, D_MODEL]
    bf16*  hbf   = (bf16*) (ws + 104 * MB);                // over xc: [FTOK, D_MODEL]
    float* gmid  = (float*)(ws + 40 * MB);                 // over xz: [FTOK, D_FF]
    bf16*  gbf   = (bf16*) (ws + 72 * MB);                 // over xz: 16 MB

    auto cvt = [&](const float* s, bf16* d, size_t n) {
        cvt_f32_bf16<<<dim3((unsigned)((n + 255) / 256)), 256, 0, stream>>>(s, d, (int)n);
    };
    auto gemm = [&](const bf16* A, const bf16* B, float* C, bf16* Cbf,
                    const float* bias, int M, int N, int K, int act) {
        dim3 grid(N / 128, M / 64);
        gemm_bf16_wmma<<<grid, 256, 0, stream>>>(A, B, C, Cbf, bias, M, N, K, act);
    };

    // 0) weight conversion
    cvt(w_in,  wb_in,  (size_t)D_MODEL * 2 * D_INNER);
    cvt_pad_xp<<<dim3((D_INNER * DBL_P + 255) / 256), 256, 0, stream>>>(w_xp, wb_xp);
    cvt(w_dt,  wb_dt,  (size_t)DT_RANK * D_INNER);
    cvt(w_out, wb_out, (size_t)D_INNER * D_MODEL);
    cvt(ff_w1, wb_f1,  (size_t)D_MODEL * D_FF);
    cvt(ff_w2, wb_f2,  (size_t)D_FF * D_MODEL);

    // 1) xx = [x ; flip(x)] -> bf16
    build_xx<<<dim3((TOK * D_MODEL + 255) / 256), 256, 0, stream>>>(x, xxbf);

    // 2) xz = xx @ in_proj_w         [4096,1024] @ [1024,4096]
    gemm(xxbf, wb_in, xz, nullptr, nullptr, TOK, 2 * D_INNER, D_MODEL, 0);

    // 3) depthwise conv + SiLU       -> xc (f32 + bf16)
    conv_silu<<<dim3((TOK * D_INNER + 255) / 256), 256, 0, stream>>>(
        xz, conv_w, conv_b, xc, xcbf);

    // 4) dbl = xc @ x_proj_w_padded  [4096,2048] @ [2048,128]
    gemm(xcbf, wb_xp, dbl, nullptr, nullptr, TOK, DBL_P, D_INNER, 0);
    extract_dt<<<dim3((TOK * DT_RANK + 255) / 256), 256, 0, stream>>>(dbl, dtbf);

    // 5) delta = softplus(dt @ dt_proj_w + b)   [4096,64] @ [64,2048]
    gemm(dtbf, wb_dt, delta, nullptr, b_dt, TOK, D_INNER, DT_RANK, 1);

    // 6) selective scan (+ skip D, * silu(z)) -> ybf
    scan_kernel<<<dim3((NB * D_INNER * D_STATE) / 256), 256, 0, stream>>>(
        xz, xc, delta, dbl, A_log, Dp, ybf);

    // 7) m = y @ out_proj_w          [4096,2048] @ [2048,1024]
    gemm(ybf, wb_out, mbuf, nullptr, nullptr, TOK, D_MODEL, D_INNER, 0);

    // 8) combine directions + LayerNorm -> hbf
    combine_ln<<<dim3(FTOK), 256, 0, stream>>>(mbuf, ln_g, ln_b, hbf);

    // 9) FFN: gelu(h @ ff_w1 + b1) -> gbf;  out = g @ ff_w2 + b2
    gemm(hbf, wb_f1, gmid, gbf, ff_b1, FTOK, D_FF, D_MODEL, 2);
    gemm(gbf, wb_f2, out, nullptr, ff_b2, FTOK, D_MODEL, D_FF, 0);
}
